// DecoderRNNWithAttention_68496138436987
// MI455X (gfx1250) — matrix operand order repaired and checked by reading, not compiled, not run on hardware
//
#include <hip/hip_runtime.h>
#include <hip/hip_bf16.h>

// ---------------------------------------------------------------------------
// Show-Attend-Tell decoder for MI455X (gfx1250, wave32, WMMA).
// Heavy GEMMs -> v_wmma_f32_16x16x32_bf16. Weights preconverted to bf16 once
// (fc_w zero-padded to a multiple of 64 rows so the GEMM inner loop is
// branch-free); per-step activations preconverted to bf16; 1x4 N-tile
// blocking per wave = 4 independent accumulators covering the WMMA->WMMA
// accumulator RAW hazard with zero EXEC manipulation in the k-loop.
// ---------------------------------------------------------------------------

#define B_  64
#define P_  196
#define ENC_ 2048
#define H_  512
#define A_  512
#define E_  512
#define V_  10000
#define VPAD 10048      // V rounded up to multiple of 64
#define TM1 19          // T-1 decode steps
#define XH  3072        // E + ENC + H (concatenated LSTM input)

typedef __bf16 bf16x16 __attribute__((ext_vector_type(16)));
typedef __bf16 bf16x8  __attribute__((ext_vector_type(8)));
typedef float  f32x8   __attribute__((ext_vector_type(8)));
typedef float  f32x4   __attribute__((ext_vector_type(4)));

__device__ __forceinline__ float sigmoidf_(float x) {
    return 1.0f / (1.0f + __expf(-x));
}

// Pack two contiguous 16-byte chunks into the WMMA 16-bit A/B operand.
__device__ __forceinline__ bf16x16 pack16(bf16x8 lo, bf16x8 hi) {
    bf16x16 r;
#pragma unroll
    for (int j = 0; j < 8; ++j) { r[j] = lo[j]; r[8 + j] = hi[j]; }
    return r;
}

// ---------------------------------------------------------------------------
// bf16-A WMMA GEMM, 1x4 N-tile blocking per wave, branch-free inner loop:
//   C[M, 0..Nout) = act( A[M,K](bf16) @ W[N,K](bf16)^T + bias )
// Requires M%16==0, N%64==0, K%32==0; columns in [Nout, N) computed but not
// stored (W must be padded/zeroed there).
// A layout (16-bit A 16x32): lane l (m=l&15) holds K = kA..kA+7 (elems 0-7)
//   and kA+16..kA+23 (elems 8-15), kA = (l>=16)*8  -> two contiguous loads.
// B layout (16-bit B 32x16): lane l (n=l&15) holds K = kB..kB+7 (elems 0-7)
//   and kB+16..kB+23 (elems 8-15), kB = (l>=16)*8 -> two contiguous loads.
// C/D: element r -> row m = 8*(l>>4)+r, col n = l&15.
// ---------------------------------------------------------------------------
__global__ __launch_bounds__(256)
void wmma_gemm_bf16a_kernel(const __bf16* __restrict__ A, int lda,
                            const __bf16* __restrict__ W, int ldw,
                            const float* __restrict__ bias,
                            float* __restrict__ C, long long ldc,
                            int M, int N, int Nout, int K, int act,
                            const int* __restrict__ lengths, int t) {
    const int lane  = threadIdx.x & 31;
    const int wave  = (blockIdx.x * blockDim.x + threadIdx.x) >> 5;
    const int nblk  = N >> 6;                        // groups of 4 N-tiles
    const int tiles = (M >> 4) * nblk;
    if (wave >= tiles) return;                       // wave-uniform

    const int tm  = (wave / nblk) << 4;
    const int tn0 = (wave % nblk) << 6;

    const int kA = (lane >> 4) << 3;                 // 0 or 8
    const int kB = (lane >> 4) << 3;                 // 0 or 8
    const __bf16* Arow = A + (long long)(tm + (lane & 15)) * lda;
    const __bf16* Wr0  = W + (long long)(tn0 +      (lane & 15)) * ldw;
    const __bf16* Wr1  = W + (long long)(tn0 + 16 + (lane & 15)) * ldw;
    const __bf16* Wr2  = W + (long long)(tn0 + 32 + (lane & 15)) * ldw;
    const __bf16* Wr3  = W + (long long)(tn0 + 48 + (lane & 15)) * ldw;

    f32x8 acc0 = {}, acc1 = {}, acc2 = {}, acc3 = {};
    for (int k0 = 0; k0 < K; k0 += 32) {
        bf16x16 amat = pack16(*(const bf16x8*)(Arow + k0 + kA),
                              *(const bf16x8*)(Arow + k0 + kA + 16));
        bf16x16 b0 = pack16(*(const bf16x8*)(Wr0 + k0 + kB),
                            *(const bf16x8*)(Wr0 + k0 + kB + 16));
        bf16x16 b1 = pack16(*(const bf16x8*)(Wr1 + k0 + kB),
                            *(const bf16x8*)(Wr1 + k0 + kB + 16));
        bf16x16 b2 = pack16(*(const bf16x8*)(Wr2 + k0 + kB),
                            *(const bf16x8*)(Wr2 + k0 + kB + 16));
        bf16x16 b3 = pack16(*(const bf16x8*)(Wr3 + k0 + kB),
                            *(const bf16x8*)(Wr3 + k0 + kB + 16));
        acc0 = __builtin_amdgcn_wmma_f32_16x16x32_bf16(false, amat, false, b0,
                                                       (short)0, acc0, false, false);
        acc1 = __builtin_amdgcn_wmma_f32_16x16x32_bf16(false, amat, false, b1,
                                                       (short)0, acc1, false, false);
        acc2 = __builtin_amdgcn_wmma_f32_16x16x32_bf16(false, amat, false, b2,
                                                       (short)0, acc2, false, false);
        acc3 = __builtin_amdgcn_wmma_f32_16x16x32_bf16(false, amat, false, b3,
                                                       (short)0, acc3, false, false);
    }

    // Epilogue: bias + activation + row masking; clamp stores to Nout.
    f32x8 accs[4] = {acc0, acc1, acc2, acc3};
#pragma unroll
    for (int i = 0; i < 4; ++i) {
        const int n = tn0 + 16 * i + (lane & 15);
        if (n >= Nout) continue;                     // per-lane, after all WMMAs
        const float bv = bias ? bias[n] : 0.0f;
#pragma unroll
        for (int r = 0; r < 8; ++r) {
            const int m = tm + ((lane >> 4) << 3) + r;
            float v = accs[i][r] + bv;
            if (act == 1) v = sigmoidf_(v);
            if (lengths && !(t < lengths[m] - 1)) v = 0.0f;
            C[(long long)m * ldc + n] = v;
        }
    }
}

// ---------------------------------------------------------------------------
// f32-A WMMA GEMM (single tile per wave) — used only for the one-time att1
// GEMM where A = encoder_out (f32, too large to be worth a bf16 copy).
// ---------------------------------------------------------------------------
__global__ __launch_bounds__(256)
void wmma_gemm_f32a_kernel(const float* __restrict__ A, int lda,
                           const __bf16* __restrict__ W, int ldw,
                           const float* __restrict__ bias,
                           float* __restrict__ C, long long ldc,
                           int M, int N, int K) {
    const int lane  = threadIdx.x & 31;
    const int wave  = (blockIdx.x * blockDim.x + threadIdx.x) >> 5;
    const int ntn   = N >> 4;
    const int tiles = (M >> 4) * ntn;
    if (wave >= tiles) return;

    const int tm = (wave / ntn) << 4;
    const int tn = (wave % ntn) << 4;
    const int kA = (lane >> 4) << 3;
    const int kB = (lane >> 4) << 3;
    const float*  Arow = A + (long long)(tm + (lane & 15)) * lda;
    const __bf16* Wrow = W + (long long)(tn + (lane & 15)) * ldw;

    f32x8 acc = {};
    for (int k0 = 0; k0 < K; k0 += 32) {
        const f32x4* ap0 = (const f32x4*)(Arow + k0 + kA);
        const f32x4* ap1 = (const f32x4*)(Arow + k0 + kA + 16);
        f32x4 a0 = ap0[0], a1 = ap0[1], a2 = ap1[0], a3 = ap1[1];
        bf16x16 amat;
#pragma unroll
        for (int j = 0; j < 4; ++j) {
            amat[j]      = (__bf16)a0[j];
            amat[4 + j]  = (__bf16)a1[j];
            amat[8 + j]  = (__bf16)a2[j];
            amat[12 + j] = (__bf16)a3[j];
        }
        bf16x16 bmat = pack16(*(const bf16x8*)(Wrow + k0 + kB),
                              *(const bf16x8*)(Wrow + k0 + kB + 16));
        acc = __builtin_amdgcn_wmma_f32_16x16x32_bf16(false, amat, false, bmat,
                                                      (short)0, acc, false, false);
    }
    const int n = tn + (lane & 15);
    const float bv = bias ? bias[n] : 0.0f;
#pragma unroll
    for (int r = 0; r < 8; ++r) {
        const int m = tm + ((lane >> 4) << 3) + r;
        C[(long long)m * ldc + n] = acc[r] + bv;
    }
}

// --------------------------- setup kernels ---------------------------------
__global__ void cvt_f32_bf16_kernel(const float* __restrict__ src,
                                    __bf16* __restrict__ dst, int n) {
    int i = blockIdx.x * blockDim.x + threadIdx.x;
    if (i < n) dst[i] = (__bf16)src[i];
}

// Convert [rows_src, ld] f32 -> [rows_dst, ld] bf16, zero-filling pad rows.
__global__ void cvt_pad_f32_bf16_kernel(const float* __restrict__ src,
                                        __bf16* __restrict__ dst,
                                        int rows_src, int rows_dst, int ld) {
    int i = blockIdx.x * blockDim.x + threadIdx.x;
    if (i >= rows_dst * ld) return;
    int row = i / ld;
    dst[i] = (row < rows_src) ? (__bf16)src[i] : (__bf16)0.0f;
}

__global__ void lstm_cat_kernel(const float* __restrict__ wih,
                                const float* __restrict__ whh,
                                __bf16* __restrict__ wcat,
                                const float* __restrict__ bih,
                                const float* __restrict__ bhh,
                                float* __restrict__ bcat) {
    int i = blockIdx.x * blockDim.x + threadIdx.x;      // 2048*3072
    if (i >= 4 * H_ * XH) return;
    int row = i / XH, col = i % XH;
    float v = (col < E_ + ENC_) ? wih[(long long)row * (E_ + ENC_) + col]
                                : whh[(long long)row * H_ + (col - E_ - ENC_)];
    wcat[i] = (__bf16)v;
    if (i < 4 * H_) bcat[i] = bih[i] + bhh[i];
}

__global__ void mean_enc_kernel(const float* __restrict__ enc,
                                __bf16* __restrict__ mean_bf) {
    int i = blockIdx.x * blockDim.x + threadIdx.x;      // 64*2048
    if (i >= B_ * ENC_) return;
    int b = i / ENC_, e = i % ENC_;
    const float* p = enc + (long long)b * P_ * ENC_ + e;
    float s = 0.f;
    for (int pp = 0; pp < P_; ++pp) s += p[(long long)pp * ENC_];
    mean_bf[i] = (__bf16)(s * (1.0f / P_));
}

// --------------------------- per-step kernels ------------------------------
// e[b,p] = sum_a relu(att1[b,p,a] + att2[b,a]) * fw[a] + fb   (one wave/row)
__global__ __launch_bounds__(256)
void escore_kernel(const float* __restrict__ att1, const float* __restrict__ att2,
                   const float* __restrict__ fw, const float* __restrict__ fb,
                   float* __restrict__ e) {
    int wave = (blockIdx.x * blockDim.x + threadIdx.x) >> 5;
    int lane = threadIdx.x & 31;
    if (wave >= B_ * P_) return;
    int b = wave / P_;
    const float* a1 = att1 + (long long)wave * A_;
    const float* a2 = att2 + (long long)b * A_;
    float s = 0.f;
    for (int a = lane; a < A_; a += 32) {
        float v = a1[a] + a2[a];
        s += (v > 0.f ? v : 0.f) * fw[a];
    }
    for (int off = 16; off > 0; off >>= 1) s += __shfl_down(s, off, 32);
    if (lane == 0) e[wave] = s + fb[0];
}

// Per-batch softmax over P=196; writes alpha (state) and masked alphas output.
__global__ __launch_bounds__(256)
void softmax_kernel(const float* __restrict__ e, float* __restrict__ alpha,
                    float* __restrict__ alphas_out,
                    const int* __restrict__ lengths, int t) {
    int b = blockIdx.x, tid = threadIdx.x;
    __shared__ float red[256];
    float v = (tid < P_) ? e[b * P_ + tid] : -1e30f;
    red[tid] = v; __syncthreads();
    for (int s = 128; s > 0; s >>= 1) {
        if (tid < s) red[tid] = fmaxf(red[tid], red[tid + s]);
        __syncthreads();
    }
    float mx = red[0]; __syncthreads();
    float ex = (tid < P_) ? __expf(v - mx) : 0.f;
    red[tid] = ex; __syncthreads();
    for (int s = 128; s > 0; s >>= 1) {
        if (tid < s) red[tid] += red[tid + s];
        __syncthreads();
    }
    float inv = 1.0f / red[0];
    if (tid < P_) {
        float al = ex * inv;
        alpha[b * P_ + tid] = al;
        bool active = t < lengths[b] - 1;
        alphas_out[((long long)b * TM1 + t) * P_ + tid] = active ? al : 0.0f;
    }
}

// awe[b,e] = sum_p enc[b,p,e]*alpha[b,p];  xh_bf[b, E_+e] = bf16(gate*awe)
__global__ void awe_x_kernel(const float* __restrict__ enc,
                             const float* __restrict__ alpha,
                             const float* __restrict__ gate,
                             __bf16* __restrict__ xh_bf) {
    int i = blockIdx.x * blockDim.x + threadIdx.x;      // 64*2048
    if (i >= B_ * ENC_) return;
    int b = i / ENC_, e = i % ENC_;
    const float* pe = enc + (long long)b * P_ * ENC_ + e;
    const float* pa = alpha + b * P_;
    float s = 0.f;
    for (int p = 0; p < P_; ++p) s += pe[(long long)p * ENC_] * pa[p];
    xh_bf[(long long)b * XH + E_ + e] = (__bf16)(gate[i] * s);
}

// xh_bf[b, 0:512] = bf16(embed_w[captions[b,t]]),  xh_bf[b, 2560:3072] = h_bf[b]
__global__ void xh_fill_kernel(const float* __restrict__ embed_w,
                               const int* __restrict__ captions,
                               const __bf16* __restrict__ h_bf,
                               __bf16* __restrict__ xh_bf, int t) {
    int i = blockIdx.x * blockDim.x + threadIdx.x;      // 64*1024
    if (i >= B_ * (E_ + H_)) return;
    int b = i / (E_ + H_), j = i % (E_ + H_);
    if (j < E_)
        xh_bf[(long long)b * XH + j] =
            (__bf16)embed_w[(long long)captions[b * 20 + t] * E_ + j];
    else
        xh_bf[(long long)b * XH + (E_ + ENC_) + (j - E_)] = h_bf[b * H_ + (j - E_)];
}

// LSTM pointwise: gates[b, 4H] split (i,f,g,o); masked state update.
// h carried as bf16 (only consumed as GEMM A-operand); c carried in f32.
__global__ void lstm_update_kernel(const float* __restrict__ gates,
                                   float* __restrict__ c_state,
                                   __bf16* __restrict__ h_bf,
                                   __bf16* __restrict__ hnew_bf,
                                   const int* __restrict__ lengths, int t) {
    int idx = blockIdx.x * blockDim.x + threadIdx.x;    // 64*512
    if (idx >= B_ * H_) return;
    int b = idx / H_, j = idx % H_;
    const float* g = gates + (long long)b * 4 * H_;
    float ig = sigmoidf_(g[j]);
    float fg = sigmoidf_(g[H_ + j]);
    float gg = tanhf(g[2 * H_ + j]);
    float og = sigmoidf_(g[3 * H_ + j]);
    float cn = fg * c_state[idx] + ig * gg;
    float hn = og * tanhf(cn);
    hnew_bf[idx] = (__bf16)hn;                          // feeds fc (masked there)
    if (t < lengths[b] - 1) { c_state[idx] = cn; h_bf[idx] = (__bf16)hn; }
}

// ---------------------------------------------------------------------------
static inline size_t align256(size_t x) { return (x + 255) & ~(size_t)255; }

extern "C" void kernel_launch(void* const* d_in, const int* in_sizes, int n_in,
                              void* d_out, int out_size, void* d_ws, size_t ws_size,
                              hipStream_t stream) {
    (void)in_sizes; (void)n_in; (void)out_size; (void)ws_size;
    const float* encoder_out = (const float*)d_in[0];
    const int*   captions    = (const int*)d_in[1];
    const int*   lengths     = (const int*)d_in[2];
    const float* enc_att_w   = (const float*)d_in[3];
    const float* enc_att_b   = (const float*)d_in[4];
    const float* hid_att_w   = (const float*)d_in[5];
    const float* hid_att_b   = (const float*)d_in[6];
    const float* full_att_w  = (const float*)d_in[7];
    const float* full_att_b  = (const float*)d_in[8];
    const float* embed_w     = (const float*)d_in[9];
    const float* lstm_w_ih   = (const float*)d_in[10];
    const float* lstm_w_hh   = (const float*)d_in[11];
    const float* lstm_b_ih   = (const float*)d_in[12];
    const float* lstm_b_hh   = (const float*)d_in[13];
    const float* init_h_w    = (const float*)d_in[14];
    const float* init_h_b    = (const float*)d_in[15];
    const float* init_c_w    = (const float*)d_in[16];
    const float* init_c_b    = (const float*)d_in[17];
    const float* fbeta_w     = (const float*)d_in[18];
    const float* fbeta_b     = (const float*)d_in[19];
    const float* fc_w        = (const float*)d_in[20];
    const float* fc_b        = (const float*)d_in[21];

    float* pred_out   = (float*)d_out;                        // [64,19,10000]
    float* alphas_out = pred_out + (long long)B_ * TM1 * V_;  // [64,19,196]

    // ---- workspace carve ----
    char* wp = (char*)d_ws;
    auto carve = [&](size_t bytes) -> void* { void* p = wp; wp += align256(bytes); return p; };
    float*  att1          = (float*) carve((size_t)B_*P_*A_*4);
    __bf16* enc_att_w_bf  = (__bf16*)carve((size_t)A_*ENC_*2);
    __bf16* hid_att_w_bf  = (__bf16*)carve((size_t)A_*H_*2);
    __bf16* fbeta_w_bf    = (__bf16*)carve((size_t)ENC_*H_*2);
    __bf16* fc_w_bf       = (__bf16*)carve((size_t)VPAD*H_*2);   // zero-padded
    __bf16* init_h_w_bf   = (__bf16*)carve((size_t)H_*ENC_*2);
    __bf16* init_c_w_bf   = (__bf16*)carve((size_t)H_*ENC_*2);
    __bf16* lstm_w_bf     = (__bf16*)carve((size_t)4*H_*XH*2);
    float*  lstm_b_cat    = (float*) carve((size_t)4*H_*4);
    __bf16* mean_bf       = (__bf16*)carve((size_t)B_*ENC_*2);
    float*  h0_f32        = (float*) carve((size_t)B_*H_*4);
    float*  c_state       = (float*) carve((size_t)B_*H_*4);
    __bf16* h_bf          = (__bf16*)carve((size_t)B_*H_*2);
    __bf16* hnew_bf       = (__bf16*)carve((size_t)B_*H_*2);
    float*  att2          = (float*) carve((size_t)B_*A_*4);
    float*  escore        = (float*) carve((size_t)B_*P_*4);
    float*  alpha         = (float*) carve((size_t)B_*P_*4);
    float*  gate          = (float*) carve((size_t)B_*ENC_*4);
    __bf16* xh_bf         = (__bf16*)carve((size_t)B_*XH*2);
    float*  gates         = (float*) carve((size_t)B_*4*H_*4);

    auto cvt = [&](const float* s, __bf16* d, int n) {
        cvt_f32_bf16_kernel<<<(n + 255) / 256, 256, 0, stream>>>(s, d, n);
    };
    auto gemm = [&](const __bf16* A, int lda, const __bf16* W, int ldw,
                    const float* bias, float* C, long long ldc,
                    int M, int N, int Nout, int K, int act, const int* len, int t) {
        int tiles  = (M / 16) * (N / 64);
        int blocks = (tiles + 7) / 8;   // 8 waves per 256-thread block
        wmma_gemm_bf16a_kernel<<<blocks, 256, 0, stream>>>(A, lda, W, ldw, bias,
                                                           C, ldc, M, N, Nout, K,
                                                           act, len, t);
    };

    // ---- setup: weight conversion, mean encoder, h0/c0, att1 ----
    cvt(enc_att_w, enc_att_w_bf, A_ * ENC_);
    cvt(hid_att_w, hid_att_w_bf, A_ * H_);
    cvt(fbeta_w,   fbeta_w_bf,   ENC_ * H_);
    cvt(init_h_w,  init_h_w_bf,  H_ * ENC_);
    cvt(init_c_w,  init_c_w_bf,  H_ * ENC_);
    {   // fc_w: 10000 -> 10048 rows, pad rows zeroed
        int n = VPAD * H_;
        cvt_pad_f32_bf16_kernel<<<(n + 255) / 256, 256, 0, stream>>>(
            fc_w, fc_w_bf, V_, VPAD, H_);
    }
    {
        int n = 4 * H_ * XH;
        lstm_cat_kernel<<<(n + 255) / 256, 256, 0, stream>>>(
            lstm_w_ih, lstm_w_hh, lstm_w_bf, lstm_b_ih, lstm_b_hh, lstm_b_cat);
    }
    mean_enc_kernel<<<(B_ * ENC_ + 255) / 256, 256, 0, stream>>>(encoder_out, mean_bf);

    // h0 / c0 from mean encoder features
    gemm(mean_bf, ENC_, init_h_w_bf, ENC_, init_h_b, h0_f32,  H_, B_, H_, H_, ENC_, 0, nullptr, 0);
    gemm(mean_bf, ENC_, init_c_w_bf, ENC_, init_c_b, c_state, H_, B_, H_, H_, ENC_, 0, nullptr, 0);
    cvt(h0_f32, h_bf, B_ * H_);
    // att1: [B*P, ENC] x [A, ENC]^T -> [B*P, A]   (f32-A variant, one-time)
    {
        int tiles  = (B_ * P_ / 16) * (A_ / 16);
        int blocks = (tiles + 7) / 8;
        wmma_gemm_f32a_kernel<<<blocks, 256, 0, stream>>>(
            encoder_out, ENC_, enc_att_w_bf, ENC_, enc_att_b, att1, A_,
            B_ * P_, A_, ENC_);
    }

    // ---- decode loop ----
    for (int t = 0; t < TM1; ++t) {
        // att2 = h @ hid_att_w^T + b          [64,512]
        gemm(h_bf, H_, hid_att_w_bf, H_, hid_att_b, att2, A_, B_, A_, A_, H_, 0, nullptr, 0);
        // e scores + softmax -> alpha (+ masked alphas out)
        escore_kernel<<<(B_ * P_ + 7) / 8, 256, 0, stream>>>(att1, att2, full_att_w, full_att_b, escore);
        softmax_kernel<<<B_, 256, 0, stream>>>(escore, alpha, alphas_out, lengths, t);
        // gate = sigmoid(h @ fbeta_w^T + b)   [64,2048]
        gemm(h_bf, H_, fbeta_w_bf, H_, fbeta_b, gate, ENC_, B_, ENC_, ENC_, H_, 1, nullptr, 0);
        // xh = [emb_t, gate*awe, h]           [64,3072] bf16
        awe_x_kernel<<<(B_ * ENC_ + 255) / 256, 256, 0, stream>>>(encoder_out, alpha, gate, xh_bf);
        xh_fill_kernel<<<(B_ * (E_ + H_) + 255) / 256, 256, 0, stream>>>(embed_w, captions, h_bf, xh_bf, t);
        // gates = xh @ [W_ih|W_hh]^T + (b_ih+b_hh)   [64,2048], K=3072
        gemm(xh_bf, XH, lstm_w_bf, XH, lstm_b_cat, gates, 4 * H_, B_, 4 * H_, 4 * H_, XH, 0, nullptr, 0);
        // LSTM pointwise + masked state update (h in bf16, c in f32)
        lstm_update_kernel<<<(B_ * H_ + 255) / 256, 256, 0, stream>>>(gates, c_state, h_bf, hnew_bf, lengths, t);
        // pred = h_new @ fc_w^T(padded) + fc_b -> masked, strided into d_out
        gemm(hnew_bf, H_, fc_w_bf, H_, fc_b, pred_out + (long long)t * V_,
             (long long)TM1 * V_, B_, VPAD, V_, H_, 0, lengths, t);
    }
}
